// TreeGRUEncoder_73400991088922
// MI455X (gfx1250) — matrix operand order repaired and checked by reading, not compiled, not run on hardware
//
#include <hip/hip_runtime.h>
#include <stddef.h>

// ---------------- problem constants ----------------
#define LSEQ 1024
#define BB   64
#define DIN  512
#define HH   512
#define G5   2560      // 5*H

// ---------------- persistent-kernel config ----------------
// 80 gate WGs (32 of 2560 cols each) + 16 cell WGs (32 of 512 cols each).
// 8 waves per WG; each wave owns one fixed 16x16 output tile.
#define NWG_GATE 80
#define NWG_CELL 16
#define NWG      (NWG_GATE + NWG_CELL)
#define TPB      256

// ---------------- workspace layout (bytes) ----------------
#define XBF_OFF   ((size_t)0)                               // L*B*DIN bf16
#define WG_OFF    (XBF_OFF  + (size_t)LSEQ*BB*DIN*2)        // 2560x512 bf16
#define WGL_OFF   (WG_OFF   + (size_t)G5*HH*2)              // 2560x512 bf16
#define WC_OFF    (WGL_OFF  + (size_t)G5*HH*2)              // 512x512 bf16
#define WCL_OFF   (WC_OFF   + (size_t)HH*HH*2)              // 512x512 bf16
#define H_OFF     (WCL_OFF  + (size_t)HH*HH*2)              // 64x512 f32
#define HBF_OFF   (H_OFF    + (size_t)BB*HH*4)              // 64x512 bf16
#define GATES_OFF (HBF_OFF  + (size_t)BB*HH*2)              // 64x2560 f32
#define U_OFF     (GATES_OFF+ (size_t)BB*G5*4)              // 64x512 bf16
#define SEM_OFF   (U_OFF    + (size_t)BB*HH*2)              // semaphore words

// async-to-LDS availability (gfx1250 builtins; fall back to plain copy)
#if defined(__has_builtin)
#if __has_builtin(__builtin_amdgcn_global_load_async_to_lds_b128) && \
    __has_builtin(__builtin_amdgcn_s_wait_asynccnt)
#define HAVE_ASYNC_LDS 1
#endif
#endif

// ---------------- types ----------------
typedef __attribute__((ext_vector_type(16))) __bf16 bf16x16;
typedef __attribute__((ext_vector_type(8)))  float  f32x8;

#if defined(HAVE_ASYNC_LDS)
// builtin expects: (int4-vector __device__(AS1)*, int4-vector AS3*, imm, imm)
typedef __attribute__((__vector_size__(4 * sizeof(int)))) int vint4;
typedef __attribute__((address_space(1))) vint4* as1_v4ptr;
typedef __attribute__((address_space(3))) vint4* as3_v4ptr;
#endif

// ---------------- helpers ----------------
__device__ __forceinline__ unsigned short f2bf(float f) {
  unsigned u = __float_as_uint(f);
  unsigned r = u + 0x7FFFu + ((u >> 16) & 1u);   // round-to-nearest-even
  return (unsigned short)(r >> 16);
}

__device__ __forceinline__ float sigmoidf_fast(float x) {
  return 1.0f / (1.0f + __expf(-x));
}

__device__ __forceinline__ float tanhf_fast(float x) {
  float e = __expf(-2.0f * fabsf(x));
  float r = (1.0f - e) / (1.0f + e);
  return copysignf(r, x);
}

__device__ __forceinline__ f32x8 zero8() {
  f32x8 z = {0.f, 0.f, 0.f, 0.f, 0.f, 0.f, 0.f, 0.f};
  return z;
}

// A fragment: 16x32 bf16 tile, ISA layout (lanes 0-15: M=lane, K {0..7,16..23};
// lanes 16-31: M=lane-16, K {8..15,24..31}) -> two 16B contiguous global loads.
__device__ __forceinline__ bf16x16 load_frag_a(const unsigned short* __restrict__ base,
                                               int ld, int lane) {
  union { uint4 q[2]; bf16x16 v; } u;
  const int r  = lane & 15;
  const int k0 = (lane & 16) ? 8 : 0;
  const unsigned short* p = base + (size_t)r * ld + k0;
  u.q[0] = *reinterpret_cast<const uint4*>(p);        // K = k0 .. k0+7
  u.q[1] = *reinterpret_cast<const uint4*>(p + 16);   // K = k0+16 .. k0+23
  return u.v;
}

// B fragment from LDS-resident weight slab [32 cols][512 K] bf16.
// lanes 0-15: N=nloc+lane, K k0+{0..7,16..23}; lanes 16-31: N=nloc+lane-16, K k0+{8..15,24..31}.
__device__ __forceinline__ bf16x16 load_frag_b_lds(const unsigned short* sW,
                                                   int nloc, int k0, int lane) {
  union { uint4 q[2]; bf16x16 v; } u;
  const int n = nloc + (lane & 15);
  const int k = k0 + ((lane & 16) ? 8 : 0);
  const unsigned short* p = sW + n * DIN + k;
  u.q[0] = *reinterpret_cast<const uint4*>(p);
  u.q[1] = *reinterpret_cast<const uint4*>(p + 16);
  return u.v;
}

__device__ __forceinline__ f32x8 wmma_bf16(bf16x16 a, bf16x16 b, f32x8 c) {
  return __builtin_amdgcn_wmma_f32_16x16x32_bf16(false, a, false, b, (short)0, c,
                                                 false, false);
}

// ---- one-directional producer->consumer sync (monotonic counters, no reset)
__device__ __forceinline__ void sem_signal(unsigned* sem) {
  __syncthreads();                    // all waves in this WG finished their stores
  if (threadIdx.x == 0) {
    __threadfence();
    __hip_atomic_fetch_add(sem, 1u, __ATOMIC_RELEASE, __HIP_MEMORY_SCOPE_AGENT);
  }
}

__device__ __forceinline__ void sem_wait(const unsigned* sem, unsigned target) {
  if (threadIdx.x == 0) {
    while (__hip_atomic_load(sem, __ATOMIC_ACQUIRE, __HIP_MEMORY_SCOPE_AGENT) < target) {
      __builtin_amdgcn_s_sleep(1);
    }
    __threadfence();
  }
  __syncthreads();
}

// ---------------- pre-pass kernels ----------------
__global__ void cvt_bf16_kernel(const float* __restrict__ src,
                                unsigned short* __restrict__ dst, int n) {
  int i = blockIdx.x * blockDim.x + threadIdx.x;
  int stride = gridDim.x * blockDim.x;
  for (; i < n; i += stride) dst[i] = f2bf(src[i]);
}

__global__ void init_state_kernel(float* __restrict__ h,
                                  unsigned short* __restrict__ hbf,
                                  unsigned* __restrict__ sem,
                                  float* __restrict__ out_tail) {
  int i = blockIdx.x * blockDim.x + threadIdx.x;
  int stride = gridDim.x * blockDim.x;
  if (i < 64) sem[i] = 0u;
  for (int j = i; j < BB * HH; j += stride) {
    h[j] = 0.0f;
    hbf[j] = 0;
    out_tail[j] = 0.0f;   // reference's output_t is zeros
  }
}

// ---------------- persistent recurrent kernel ----------------
// Per step t (one wait per WG group):
//   gate WGs: wait semB>=16t -> gates = sigmoid(acc_gx + h@Wgl^T + bg), u = rl*h
//             -> signal semA -> acc_gx = x_{t+1}@Wg^T (off critical path)
//   cell WGs: acc_cx = x_t@Wc^T (off critical path) -> wait semA>=80(t+1)
//             -> cell = tanh(acc_cx + u@Wcl^T + bc); h = zl*h + z*cell -> signal semB
__global__ void __launch_bounds__(TPB)
tree_gru_persistent(const unsigned short* __restrict__ Xbf,   // [L,B,DIN]
                    const unsigned short* __restrict__ Wg,    // [2560,512]
                    const unsigned short* __restrict__ Wgl,   // [2560,512]
                    const unsigned short* __restrict__ Wc,    // [512,512]
                    const unsigned short* __restrict__ Wcl,   // [512,512]
                    const float* __restrict__ bg,             // [2560]
                    const float* __restrict__ bc,             // [512]
                    float* __restrict__ h,                    // [64,512]
                    unsigned short* __restrict__ hbf,         // [64,512]
                    float* __restrict__ gates,                // [64,2560]
                    unsigned short* __restrict__ ubuf,        // [64,512]
                    float* __restrict__ out,                  // [B,L,H]
                    unsigned* __restrict__ sem) {
  __shared__ unsigned short sWx[32 * DIN];   // x-side weight slab (32 cols)
  __shared__ unsigned short sWh[32 * DIN];   // h-side weight slab (32 cols)

  unsigned* semA = sem;        // gate window-1 completions
  unsigned* semB = sem + 16;   // cell window-2 completions (separate cacheline)

  const int lane = threadIdx.x & 31;
  const int wv   = threadIdx.x >> 5;
  const bool is_gate = (int)blockIdx.x < NWG_GATE;
  const int grp  = is_gate ? (int)blockIdx.x : ((int)blockIdx.x - NWG_GATE);
  const int col0 = grp * 32;             // base output column within its matrix
  const int nloc = (wv & 1) * 16;        // local N-tile (0 or 16) in LDS slab
  const int tm   = wv >> 1;              // M tile 0..3 (B=64)

  // ---- stage this WG's weight slab into LDS once (resident for all 1024 steps)
  {
    const uint4* gx = (const uint4*)((is_gate ? Wg  : Wc)  + (size_t)col0 * DIN);
    const uint4* gh = (const uint4*)((is_gate ? Wgl : Wcl) + (size_t)col0 * DIN);
    uint4* dx = (uint4*)sWx;
    uint4* dh = (uint4*)sWh;
#if defined(HAVE_ASYNC_LDS)
    for (int i = threadIdx.x; i < (32 * DIN) / 8; i += TPB) {
      __builtin_amdgcn_global_load_async_to_lds_b128(
          (as1_v4ptr)(gx + i), (as3_v4ptr)(dx + i), 0, 0);
      __builtin_amdgcn_global_load_async_to_lds_b128(
          (as1_v4ptr)(gh + i), (as3_v4ptr)(dh + i), 0, 0);
    }
    __builtin_amdgcn_s_wait_asynccnt(0);
#else
    for (int i = threadIdx.x; i < (32 * DIN) / 8; i += TPB) {
      dx[i] = gx[i];
      dh[i] = gh[i];
    }
#endif
  }
  __syncthreads();

  const int n_out  = col0 + nloc + (lane & 15);       // fixed output column
  const int mbase  = tm * 16 + ((lane & 16) ? 8 : 0); // fixed row base
  const float bias = is_gate ? bg[n_out] : bc[n_out];

  f32x8 acc = zero8();

  // ---- prologue: gate WGs pre-accumulate x-part for t = 0
  if (is_gate) {
    const unsigned short* arow = Xbf + (size_t)(tm * 16) * DIN;
    #pragma unroll 4
    for (int kt = 0; kt < DIN; kt += 32)
      acc = wmma_bf16(load_frag_a(arow + kt, DIN, lane),
                      load_frag_b_lds(sWx, nloc, kt, lane), acc);
  }

  for (int t = 0; t < LSEQ; ++t) {
    if (is_gate) {
      // -------- critical window: need h_{t-1} (cell step t-1 done) --------
      sem_wait(semB, (unsigned)(NWG_CELL * t));
      const unsigned short* arow_h = hbf + (size_t)(tm * 16) * HH;
      #pragma unroll 4
      for (int kt = 0; kt < HH; kt += 32)
        acc = wmma_bf16(load_frag_a(arow_h + kt, HH, lane),
                        load_frag_b_lds(sWh, nloc, kt, lane), acc);
      #pragma unroll
      for (int j = 0; j < 8; ++j) {
        const int m = mbase + j;
        const float g = sigmoidf_fast(acc[j] + bias);
        gates[m * G5 + n_out] = g;
        if (n_out < HH)                         // rl region: fuse u = rl * h_prev
          ubuf[m * HH + n_out] = f2bf(g * h[m * HH + n_out]);
      }
      sem_signal(semA);                         // gates & u ready for cell WGs

      // -------- hidden window: gate x-part for t+1 (no dependency) --------
      acc = zero8();
      if (t + 1 < LSEQ) {
        const unsigned short* arow_x = Xbf + ((size_t)(t + 1) * BB + tm * 16) * DIN;
        #pragma unroll 4
        for (int kt = 0; kt < DIN; kt += 32)
          acc = wmma_bf16(load_frag_a(arow_x + kt, DIN, lane),
                          load_frag_b_lds(sWx, nloc, kt, lane), acc);
      }
    } else {
      // -------- hidden window: cell x-part for t (no dependency) --------
      acc = zero8();
      const unsigned short* arow_x = Xbf + ((size_t)t * BB + tm * 16) * DIN;
      #pragma unroll 4
      for (int kt = 0; kt < DIN; kt += 32)
        acc = wmma_bf16(load_frag_a(arow_x + kt, DIN, lane),
                        load_frag_b_lds(sWx, nloc, kt, lane), acc);

      // -------- critical window: need gates & u of step t --------
      sem_wait(semA, (unsigned)(NWG_GATE * (t + 1)));
      const unsigned short* arow_u = ubuf + (size_t)(tm * 16) * HH;
      #pragma unroll 4
      for (int kt = 0; kt < HH; kt += 32)
        acc = wmma_bf16(load_frag_a(arow_u + kt, HH, lane),
                        load_frag_b_lds(sWh, nloc, kt, lane), acc);
      #pragma unroll
      for (int j = 0; j < 8; ++j) {
        const int m    = mbase + j;
        const float cv = tanhf_fast(acc[j] + bias);
        const float zl = gates[m * G5 + 2 * HH + n_out];  // split: [rl rr zl zr z]
        const float zz = gates[m * G5 + 4 * HH + n_out];
        const float hv = zl * h[m * HH + n_out] + zz * cv; // right child is zero
        h[m * HH + n_out]   = hv;                          // unique (m,n) owner
        hbf[m * HH + n_out] = f2bf(hv);
        out[(size_t)m * LSEQ * HH + (size_t)t * HH + n_out] = hv;
      }
      sem_signal(semB);                         // h_t ready for gate WGs
    }
  }
}

// ---------------- host launcher ----------------
extern "C" void kernel_launch(void* const* d_in, const int* in_sizes, int n_in,
                              void* d_out, int out_size, void* d_ws, size_t ws_size,
                              hipStream_t stream) {
  (void)in_sizes; (void)n_in; (void)out_size; (void)ws_size;

  const float* X     = (const float*)d_in[0];
  // d_in[1] = indexes (unused: canonical chain ordering)
  const float* Wgih  = (const float*)d_in[2];
  const float* bg    = (const float*)d_in[3];
  const float* Wglhh = (const float*)d_in[4];
  // d_in[5] = W_grhh (unused: right child h == 0)
  const float* Wcih  = (const float*)d_in[6];
  const float* bc    = (const float*)d_in[7];
  const float* Wclhh = (const float*)d_in[8];
  // d_in[9] = W_crhh (unused)

  char* ws = (char*)d_ws;
  unsigned short* xbf = (unsigned short*)(ws + XBF_OFF);
  unsigned short* wg  = (unsigned short*)(ws + WG_OFF);
  unsigned short* wgl = (unsigned short*)(ws + WGL_OFF);
  unsigned short* wc  = (unsigned short*)(ws + WC_OFF);
  unsigned short* wcl = (unsigned short*)(ws + WCL_OFF);
  float*          h   = (float*)(ws + H_OFF);
  unsigned short* hbf = (unsigned short*)(ws + HBF_OFF);
  float*          gts = (float*)(ws + GATES_OFF);
  unsigned short* ub  = (unsigned short*)(ws + U_OFF);
  unsigned*       sem = (unsigned*)(ws + SEM_OFF);

  float* out      = (float*)d_out;
  float* out_tail = out + (size_t)BB * LSEQ * HH;   // output_t zeros

  // bf16 conversion pre-passes
  cvt_bf16_kernel<<<2048, 256, 0, stream>>>(X,     xbf, LSEQ * BB * DIN);
  cvt_bf16_kernel<<<512,  256, 0, stream>>>(Wgih,  wg,  G5 * DIN);
  cvt_bf16_kernel<<<512,  256, 0, stream>>>(Wglhh, wgl, G5 * HH);
  cvt_bf16_kernel<<<128,  256, 0, stream>>>(Wcih,  wc,  HH * DIN);
  cvt_bf16_kernel<<<128,  256, 0, stream>>>(Wclhh, wcl, HH * HH);

  // zero h state, semaphores, and output_t tail (workspace is poisoned)
  init_state_kernel<<<64, 256, 0, stream>>>(h, hbf, sem, out_tail);

  // persistent recurrent kernel: LDS-resident weights, semaphore handoff
  tree_gru_persistent<<<NWG, TPB, 0, stream>>>(xbf, wg, wgl, wc, wcl, bg, bc,
                                               h, hbf, gts, ub, out, sem);
}
